// Net_36180804502142
// MI455X (gfx1250) — compile-verified, matrix operand order checked by reading, not run on hardware
//
#include <hip/hip_runtime.h>

// GatedGCN (N=50k nodes, E=1.6M edges, 32ch, 12 layers) for MI455X gfx1250.
// Memory-bound (~1 GB/layer streaming e/edge_in; node tables L2-resident).
// Dense 32x32 transforms: v_wmma_f32_16x16x32_bf16 (K=32 == channel dim).
// Tile staging: global_load_async_to_lds_b128 (ASYNCcnt) when available.

#define UNITS 32
#define DEPTH 12
#define EPSV 1e-5f

typedef __bf16 bf16_t;
typedef __attribute__((ext_vector_type(16))) __bf16 v16bf;
typedef __attribute__((ext_vector_type(8)))  float  v8f;

#if defined(__has_builtin)
#if __has_builtin(__builtin_amdgcn_global_load_async_to_lds_b128) && \
    __has_builtin(__builtin_amdgcn_s_wait_asynccnt)
#define HAVE_ASYNC_LDS 1
#endif
#endif
#ifndef HAVE_ASYNC_LDS
#define HAVE_ASYNC_LDS 0
#endif

#if HAVE_ASYNC_LDS
// Builtin signature (from hipcc diagnostic): param0 = int4 addrspace(1)*,
// param1 = int4 addrspace(3)*, then imm offset, imm cpol.
typedef int v4i32 __attribute__((vector_size(16)));
typedef __attribute__((address_space(1))) v4i32 gv4i;
typedef __attribute__((address_space(3))) v4i32 lv4i;
#endif

// Native f32->bf16 convert (single v_cvt on gfx1250)
__device__ __forceinline__ bf16_t to_bf16(float f) { return (bf16_t)f; }

// Force a predicate to be treated as wave-uniform (it is by construction):
// turns guarded fast paths into scalar s_cbranch instead of exec-mask churn.
__device__ __forceinline__ bool wave_uniform(bool p) {
  return __builtin_amdgcn_readfirstlane((int)p) != 0;
}

// ISA 7.12.2: 16-bit A 16x32 — lanes 0-15 hold K={0..7,16..23}, lanes 16-31 K={8..15,24..31}
__device__ __forceinline__ void kmap(int lane, int* ks) {
  int hi = (lane >> 4) & 1;
#pragma unroll
  for (int i = 0; i < 8; ++i) { ks[i] = 8 * hi + i; ks[i + 8] = 16 + 8 * hi + i; }
}

// A-fragment from an f32 LDS tile (convert to bf16 at build time)
__device__ __forceinline__ v16bf load_a_f32(const float* tile, int lane) {
  int ks[16]; kmap(lane, ks);
  int m = lane & 15;
  v16bf a;
#pragma unroll
  for (int i = 0; i < 16; ++i) a[i] = to_bf16(tile[m * UNITS + ks[i]]);
  return a;
}

// W stored row-major [n_out][k_in] as bf16; B(k,n) = W[n][k] (computes x @ W^T)
__device__ __forceinline__ v16bf load_b(const bf16_t* W, int lane, int nh) {
  int ks[16]; kmap(lane, ks);
  int n = (lane & 15) + 16 * nh;
  v16bf b;
#pragma unroll
  for (int i = 0; i < 16; ++i) b[i] = W[n * UNITS + ks[i]];
  return b;
}

__device__ __forceinline__ v8f wmma_bf16(v16bf a, v16bf b, v8f c) {
  return __builtin_amdgcn_wmma_f32_16x16x32_bf16(false, a, false, b, (short)0, c,
                                                 false, false);
}

// Stage a 16x32 f32 tile into LDS. Full tiles are contiguous 2KB: async DMA to LDS.
// Tail tiles: clamped synchronous float4 loads.
__device__ __forceinline__ void stage_tile16(float* tile, const float* __restrict__ src,
                                             long base, long nrows, int lane) {
  if (wave_uniform(base + 16 <= nrows)) {
#if HAVE_ASYNC_LDS
    const float* g = src + base * UNITS + lane * 4;
    float* l = tile + lane * 4;
#pragma unroll
    for (int i = 0; i < 4; ++i)
      __builtin_amdgcn_global_load_async_to_lds_b128(
          (gv4i*)(g + i * 128), (lv4i*)(l + i * 128), 0, 0);
#else
    int m = lane & 15, half = lane >> 4;
    const float4* p = (const float4*)(src + (base + m) * UNITS + half * 16);
    float4* d = (float4*)(tile + m * UNITS + half * 16);
    d[0] = p[0]; d[1] = p[1]; d[2] = p[2]; d[3] = p[3];
#endif
  } else {
    int m = lane & 15, half = lane >> 4;
    long r = base + m; if (r >= nrows) r = nrows - 1; if (r < 0) r = 0;
    const float4* p = (const float4*)(src + r * UNITS + half * 16);
    float4* d = (float4*)(tile + m * UNITS + half * 16);
    d[0] = p[0]; d[1] = p[1]; d[2] = p[2]; d[3] = p[3];
  }
}

__device__ __forceinline__ void stage_fence() {
#if HAVE_ASYNC_LDS
  __builtin_amdgcn_s_wait_asynccnt(0);
#endif
  __syncthreads();
}

__device__ __forceinline__ float fsigmoid(float x) { return 1.f / (1.f + __expf(-x)); }

// ---------------- init embeddings: out[i,c] = silu(x[i]*w[c]+b[c]) ----------------
__global__ void __launch_bounds__(256)
init_embed_kernel(const float* __restrict__ x, const float* __restrict__ w,
                  const float* __restrict__ b, float* __restrict__ out, int total) {
  int i = blockIdx.x * 256 + threadIdx.x;
  if (i >= total) return;
  int r = i >> 5, c = i & 31;
  float z = x[r] * w[c] + b[c];
  out[i] = z * fsigmoid(z);
}

// ---------------- node linears: ux/vx/bx/cx = h @ {u,v,B,C}^T + bias ----------------
__global__ void __launch_bounds__(128)
node_linear_kernel(const float* __restrict__ h,
                   const float* __restrict__ uw, const float* __restrict__ ub,
                   const float* __restrict__ vw, const float* __restrict__ vb,
                   const float* __restrict__ bw, const float* __restrict__ bb,
                   const float* __restrict__ cw, const float* __restrict__ cb,
                   float* __restrict__ ux, float* __restrict__ vx,
                   float* __restrict__ bx, float* __restrict__ cx, int n) {
  __shared__ bf16_t W[4][UNITS * UNITS];
  __shared__ float  tile[4][16 * UNITS];
  int tid = threadIdx.x, w = tid >> 5, lane = tid & 31;
  for (int i = tid; i < UNITS * UNITS; i += 128) {
    W[0][i] = to_bf16(uw[i]); W[1][i] = to_bf16(vw[i]);
    W[2][i] = to_bf16(bw[i]); W[3][i] = to_bf16(cw[i]);
  }
  long base = (long)blockIdx.x * 64 + w * 16;
  stage_tile16(tile[w], h, base, n, lane);
  stage_fence();
  v16bf a = load_a_f32(tile[w], lane);
  const float* biases[4] = {ub, vb, bb, cb};
  float* outs[4] = {ux, vx, bx, cx};
  int mhi = (lane >> 4) * 8;
  bool full = wave_uniform(base + 16 <= (long)n);
#pragma unroll
  for (int mat = 0; mat < 4; ++mat) {
#pragma unroll
    for (int nh = 0; nh < 2; ++nh) {
      v16bf bfrag = load_b(W[mat], lane, nh);
      int N = (lane & 15) + 16 * nh;
      float bias = biases[mat][N];
      v8f c;
#pragma unroll
      for (int r = 0; r < 8; ++r) c[r] = bias;
      c = wmma_bf16(a, bfrag, c);
      int rb = (int)base + mhi;
      float* p = outs[mat] + rb * UNITS + N;  // voffset once; r*128B folds to imm
      if (full) {
#pragma unroll
        for (int r = 0; r < 8; ++r) p[r * UNITS] = c[r];
      } else {
#pragma unroll
        for (int r = 0; r < 8; ++r)
          if (rb + r < n) p[r * UNITS] = c[r];
      }
    }
  }
}

// ------- edge transform: edge_in = e@A^T + ab + bx[row] + cx[col]; BN stats -------
__global__ void __launch_bounds__(256)
edge_transform_kernel(const float* __restrict__ e,
                      const int* __restrict__ row, const int* __restrict__ col,
                      const float* __restrict__ aw, const float* __restrict__ ab,
                      const float* __restrict__ bx, const float* __restrict__ cx,
                      float* __restrict__ edge_in, float* __restrict__ stat,
                      long nedges) {
  __shared__ bf16_t W[UNITS * UNITS];
  __shared__ float  tile[8][16 * UNITS];
  __shared__ float  ssum[64];
  int tid = threadIdx.x, w = tid >> 5, lane = tid & 31;
  for (int i = tid; i < UNITS * UNITS; i += 256) W[i] = to_bf16(aw[i]);
  if (tid < 64) ssum[tid] = 0.f;
  long base = (long)blockIdx.x * 128 + w * 16;
  // hint next block's tile toward the caches (global_prefetch_b8)
  long pre = base + 128; if (pre >= nedges) pre = nedges - 1;
  __builtin_prefetch(e + pre * UNITS, 0, 0);
  stage_tile16(tile[w], e, base, nedges, lane);
  stage_fence();
  v16bf a = load_a_f32(tile[w], lane);
  int mhi = (lane >> 4) * 8;
  bool full = wave_uniform(base + 16 <= nedges);
  int ridx[8], cidx[8];
  int eb = (int)base + mhi;
  if (full) {
#pragma unroll
    for (int r = 0; r < 8; ++r) { ridx[r] = row[eb + r]; cidx[r] = col[eb + r]; }
  } else {
#pragma unroll
    for (int r = 0; r < 8; ++r) {
      long eidx = base + mhi + r; if (eidx >= nedges) eidx = nedges - 1;
      ridx[r] = row[eidx]; cidx[r] = col[eidx];
    }
  }
#pragma unroll
  for (int nh = 0; nh < 2; ++nh) {
    v16bf bfrag = load_b(W, lane, nh);
    int N = (lane & 15) + 16 * nh;
    float bias = ab[N];
    v8f c;
#pragma unroll
    for (int r = 0; r < 8; ++r) c[r] = bias;
    c = wmma_bf16(a, bfrag, c);
    float lsum = 0.f, lsq = 0.f;
    float* p = edge_in + eb * UNITS + N;
    if (full) {
#pragma unroll
      for (int r = 0; r < 8; ++r) {
        float v = c[r] + bx[ridx[r] * UNITS + N] + cx[cidx[r] * UNITS + N];
        p[r * UNITS] = v;
        lsum += v; lsq += v * v;
      }
    } else {
#pragma unroll
      for (int r = 0; r < 8; ++r) {
        if (base + mhi + r < nedges) {
          float v = c[r] + bx[ridx[r] * UNITS + N] + cx[cidx[r] * UNITS + N];
          p[r * UNITS] = v;
          lsum += v; lsq += v * v;
        }
      }
    }
    atomicAdd(&ssum[N], lsum);
    atomicAdd(&ssum[32 + N], lsq);
  }
  __syncthreads();
  if (tid < 64) atomicAdd(&stat[tid], ssum[tid]);
}

// ---------------- BN finalize: scale/shift per channel ----------------
__global__ void bn_finalize_kernel(const float* __restrict__ stat,
                                   const float* __restrict__ g, const float* __restrict__ b,
                                   float* __restrict__ sc, float* __restrict__ sh,
                                   float count) {
  int c = threadIdx.x;  // 32 threads
  float mu = stat[c] / count;
  float var = stat[32 + c] / count - mu * mu;
  float s = g[c] * rsqrtf(var + EPSV);
  sc[c] = s; sh[c] = b[c] - mu * s;
}

// ------- edge update + gate + scatter: e += relu(bn); agg[row] += sig(e)*vx[col] -------
__global__ void __launch_bounds__(256)
edge_update_agg_kernel(float* __restrict__ e, const float* __restrict__ edge_in,
                       const float* __restrict__ sc, const float* __restrict__ sh,
                       const int* __restrict__ row, const int* __restrict__ col,
                       const float* __restrict__ vx, float* __restrict__ agg,
                       int total) {
  int i = blockIdx.x * 256 + threadIdx.x;
  if (i >= total) return;
  int c = i & 31;
  int eidx = i >> 5;
  float v = edge_in[i] * sc[c] + sh[c];
  float en = e[i] + fmaxf(v, 0.f);
  e[i] = en;
  float contrib = fsigmoid(en) * vx[col[eidx] * UNITS + c];
  atomicAdd(&agg[row[eidx] * UNITS + c], contrib);  // global_atomic_add_f32
}

// ---------------- node_in = ux + agg; accumulate node BN stats ----------------
__global__ void __launch_bounds__(256)
node_stats_kernel(const float* __restrict__ ux, const float* __restrict__ agg,
                  float* __restrict__ node_in, float* __restrict__ stat, int total) {
  __shared__ float ssum[64];
  int tid = threadIdx.x;
  if (tid < 64) ssum[tid] = 0.f;
  __syncthreads();
  int c = tid & 31;  // stride is a multiple of 32 -> channel fixed per thread
  float lsum = 0.f, lsq = 0.f;
  for (int i = blockIdx.x * 256 + tid; i < total; i += gridDim.x * 256) {
    float v = ux[i] + agg[i];
    node_in[i] = v;
    lsum += v; lsq += v * v;
  }
  atomicAdd(&ssum[c], lsum);
  atomicAdd(&ssum[32 + c], lsq);
  __syncthreads();
  if (tid < 64) atomicAdd(&stat[tid], ssum[tid]);
}

// ---------------- h += relu(bn(node_in)) ----------------
__global__ void __launch_bounds__(256)
node_update_kernel(float* __restrict__ h, const float* __restrict__ node_in,
                   const float* __restrict__ sc, const float* __restrict__ sh,
                   int total) {
  int i = blockIdx.x * 256 + threadIdx.x;
  if (i >= total) return;
  int c = i & 31;
  h[i] += fmaxf(node_in[i] * sc[c] + sh[c], 0.f);
}

// ------- head: z = sigmoid(silu(silu(e@W0^T+b0)@W1^T+b1)@w2^T+b2), WMMA x2 -------
__global__ void __launch_bounds__(128)
head_kernel(const float* __restrict__ e,
            const float* __restrict__ w0, const float* __restrict__ b0,
            const float* __restrict__ w1, const float* __restrict__ b1,
            const float* __restrict__ w2, const float* __restrict__ b2,
            float* __restrict__ out, long nedges) {
  __shared__ bf16_t W0[UNITS * UNITS], W1[UNITS * UNITS];
  __shared__ float  tile[4][16 * UNITS];
  __shared__ float  zfin[4][16 * UNITS];
  int tid = threadIdx.x, w = tid >> 5, lane = tid & 31;
  for (int i = tid; i < UNITS * UNITS; i += 128) { W0[i] = to_bf16(w0[i]); W1[i] = to_bf16(w1[i]); }
  long base = (long)blockIdx.x * 64 + w * 16;
  stage_tile16(tile[w], e, base, nedges, lane);
  stage_fence();
  v16bf a = load_a_f32(tile[w], lane);
  int mhi = (lane >> 4) * 8;
  // layer 0 -> silu -> back into own f32 tile
  float h0[2][8];
#pragma unroll
  for (int nh = 0; nh < 2; ++nh) {
    v16bf bfrag = load_b(W0, lane, nh);
    int N = (lane & 15) + 16 * nh;
    float bias = b0[N];
    v8f c;
#pragma unroll
    for (int r = 0; r < 8; ++r) c[r] = bias;
    c = wmma_bf16(a, bfrag, c);
#pragma unroll
    for (int r = 0; r < 8; ++r) { float v = c[r]; h0[nh][r] = v * fsigmoid(v); }
  }
  __syncthreads();
#pragma unroll
  for (int nh = 0; nh < 2; ++nh) {
    int N = (lane & 15) + 16 * nh;
#pragma unroll
    for (int r = 0; r < 8; ++r) tile[w][(mhi + r) * UNITS + N] = h0[nh][r];
  }
  __syncthreads();
  v16bf a1 = load_a_f32(tile[w], lane);
#pragma unroll
  for (int nh = 0; nh < 2; ++nh) {
    v16bf bfrag = load_b(W1, lane, nh);
    int N = (lane & 15) + 16 * nh;
    float bias = b1[N];
    v8f c;
#pragma unroll
    for (int r = 0; r < 8; ++r) c[r] = bias;
    c = wmma_bf16(a1, bfrag, c);
#pragma unroll
    for (int r = 0; r < 8; ++r) { float v = c[r]; zfin[w][(mhi + r) * UNITS + N] = v * fsigmoid(v); }
  }
  __syncthreads();
  if (lane < 16) {
    long eidx = base + lane;
    if (eidx < nedges) {
      float acc = b2[0];
#pragma unroll
      for (int k = 0; k < UNITS; ++k) acc += zfin[w][lane * UNITS + k] * w2[k];
      out[eidx] = fsigmoid(acc);
    }
  }
}

extern "C" void kernel_launch(void* const* d_in, const int* in_sizes, int n_in,
                              void* d_out, int out_size, void* d_ws, size_t ws_size,
                              hipStream_t stream) {
  const float* x    = (const float*)d_in[0];
  const int*   ei   = (const int*)d_in[1];
  const float* ea   = (const float*)d_in[2];
  const float* v0w  = (const float*)d_in[3];
  const float* v0b  = (const float*)d_in[4];
  const float* e0w  = (const float*)d_in[5];
  const float* e0b  = (const float*)d_in[6];
  const float* u_w  = (const float*)d_in[7];  const float* u_b = (const float*)d_in[8];
  const float* v_w  = (const float*)d_in[9];  const float* v_b = (const float*)d_in[10];
  const float* A_w  = (const float*)d_in[11]; const float* A_b = (const float*)d_in[12];
  const float* B_w  = (const float*)d_in[13]; const float* B_b = (const float*)d_in[14];
  const float* C_w  = (const float*)d_in[15]; const float* C_b = (const float*)d_in[16];
  const float* bn_e_g = (const float*)d_in[17]; const float* bn_e_b = (const float*)d_in[18];
  const float* bn_n_g = (const float*)d_in[19]; const float* bn_n_b = (const float*)d_in[20];
  const float* pw0 = (const float*)d_in[21]; const float* pb0 = (const float*)d_in[22];
  const float* pw1 = (const float*)d_in[23]; const float* pb1 = (const float*)d_in[24];
  const float* pw2 = (const float*)d_in[25]; const float* pb2 = (const float*)d_in[26];

  const long N = in_sizes[0];
  const long E = in_sizes[2];
  const int* row = ei;
  const int* col = ei + E;

  float* ws = (float*)d_ws;
  float* h       = ws; ws += N * UNITS;
  float* e       = ws; ws += E * UNITS;
  float* ux      = ws; ws += N * UNITS;
  float* vx      = ws; ws += N * UNITS;
  float* bx      = ws; ws += N * UNITS;
  float* cx      = ws; ws += N * UNITS;
  float* agg     = ws; ws += N * UNITS;
  float* node_in = ws; ws += N * UNITS;
  float* edge_in = ws; ws += E * UNITS;
  float* estat   = ws; ws += 64;
  float* nstat   = ws; ws += 64;
  float* esc     = ws; ws += 32;
  float* esh     = ws; ws += 32;
  float* nsc     = ws; ws += 32;
  float* nsh     = ws; ws += 32;

  init_embed_kernel<<<(unsigned)((N * UNITS + 255) / 256), 256, 0, stream>>>(
      x, v0w, v0b, h, (int)(N * UNITS));
  init_embed_kernel<<<(unsigned)((E * UNITS + 255) / 256), 256, 0, stream>>>(
      ea, e0w, e0b, e, (int)(E * UNITS));

  for (int l = 0; l < DEPTH; ++l) {
    const float* uw = u_w + (long)l * UNITS * UNITS; const float* ub = u_b + (long)l * UNITS;
    const float* vw = v_w + (long)l * UNITS * UNITS; const float* vb = v_b + (long)l * UNITS;
    const float* aw = A_w + (long)l * UNITS * UNITS; const float* ab = A_b + (long)l * UNITS;
    const float* bw = B_w + (long)l * UNITS * UNITS; const float* bb = B_b + (long)l * UNITS;
    const float* cw = C_w + (long)l * UNITS * UNITS; const float* cb = C_b + (long)l * UNITS;

    (void)hipMemsetAsync(agg,   0, (size_t)N * UNITS * sizeof(float), stream);
    (void)hipMemsetAsync(estat, 0, 64 * sizeof(float), stream);
    (void)hipMemsetAsync(nstat, 0, 64 * sizeof(float), stream);

    node_linear_kernel<<<(unsigned)((N + 63) / 64), 128, 0, stream>>>(
        h, uw, ub, vw, vb, bw, bb, cw, cb, ux, vx, bx, cx, (int)N);

    edge_transform_kernel<<<(unsigned)((E + 127) / 128), 256, 0, stream>>>(
        e, row, col, aw, ab, bx, cx, edge_in, estat, E);

    bn_finalize_kernel<<<1, 32, 0, stream>>>(estat, bn_e_g + (long)l * UNITS,
                                             bn_e_b + (long)l * UNITS, esc, esh, (float)E);

    edge_update_agg_kernel<<<(unsigned)((E * UNITS + 255) / 256), 256, 0, stream>>>(
        e, edge_in, esc, esh, row, col, vx, agg, (int)(E * UNITS));

    node_stats_kernel<<<1024, 256, 0, stream>>>(ux, agg, node_in, nstat, (int)(N * UNITS));

    bn_finalize_kernel<<<1, 32, 0, stream>>>(nstat, bn_n_g + (long)l * UNITS,
                                             bn_n_b + (long)l * UNITS, nsc, nsh, (float)N);

    node_update_kernel<<<(unsigned)((N * UNITS + 255) / 256), 256, 0, stream>>>(
        h, node_in, nsc, nsh, (int)(N * UNITS));
  }

  head_kernel<<<(unsigned)((E + 63) / 64), 128, 0, stream>>>(
      e, pw0, pb0, pw1, pb1, pw2, pb2, (float*)d_out, E);
}
